// PhiMambaDecoderLayer_27882927685973
// MI455X (gfx1250) — compile-verified, matrix operand order checked by reading, not run on hardware
//
#include <hip/hip_runtime.h>
#include <hip/hip_bf16.h>
#include <math.h>

// ---------------- problem constants ----------------
#define BATCH   2
#define SEQ     2048
#define DMODEL  2048
#define DINNER  4096
#define DXB     1024
#define HEADDIM 64
#define DSTATE  64
#define NHEADS  64
#define NKV     16
#define INTER   8192
#define NPROJ   10304        // D_IN_PROJ = 4096+1024+1024+4096+64
#define NPROJ_PAD 10496      // 41 * 256
#define NTOK    (BATCH*SEQ)  // 4096 tokens
#define EPSV    1e-5f

// offsets inside a zxbcdt row
#define OFF_Z   0
#define OFF_X   4096
#define OFF_B   5120
#define OFF_C   6144
#define OFF_DT  10240

typedef __attribute__((ext_vector_type(16))) __bf16 v16bf;
typedef __attribute__((ext_vector_type(8)))  __bf16 v8bf;
typedef __attribute__((ext_vector_type(8)))  float  v8f;

// ---------------- helpers ----------------
__device__ inline v16bf ldfrag(const __bf16* p) {
    // CDNA5 16-bit operand fragment: halves {k..k+7} and {k+16..k+23}
    v8bf lo = *(const v8bf*)(p);
    v8bf hi = *(const v8bf*)(p + 16);
    v16bf r;
#pragma unroll
    for (int i = 0; i < 8; ++i) { r[i] = lo[i]; r[i + 8] = hi[i]; }
    return r;
}

__device__ inline float gelu_tanh(float x) {
    float x3 = x * x * x;
    return 0.5f * x * (1.0f + tanhf(0.7978845608028654f * (x + 0.044715f * x3)));
}

// ---------------- layernorm -> bf16 activations ----------------
__global__ __launch_bounds__(256)
void layernorm_to_bf16(const float* __restrict__ x,
                       const float* __restrict__ w,
                       const float* __restrict__ b,
                       __bf16* __restrict__ out) {
    __shared__ float red[256];
    const long t = blockIdx.x;                 // token
    const float* xr = x + t * (long)DMODEL;
    float s = 0.f, ss = 0.f;
    for (int i = threadIdx.x; i < DMODEL; i += 256) {
        float v = xr[i]; s += v; ss += v * v;
    }
    red[threadIdx.x] = s; __syncthreads();
    for (int st = 128; st > 0; st >>= 1) {
        if (threadIdx.x < st) red[threadIdx.x] += red[threadIdx.x + st];
        __syncthreads();
    }
    const float mean = red[0] * (1.0f / DMODEL);
    __syncthreads();
    red[threadIdx.x] = ss; __syncthreads();
    for (int st = 128; st > 0; st >>= 1) {
        if (threadIdx.x < st) red[threadIdx.x] += red[threadIdx.x + st];
        __syncthreads();
    }
    const float var = red[0] * (1.0f / DMODEL) - mean * mean;
    const float rs  = rsqrtf(var + EPSV);
    for (int i = threadIdx.x; i < DMODEL; i += 256) {
        float v = (xr[i] - mean) * rs * w[i] + b[i];
        out[t * (long)DMODEL + i] = (__bf16)v;
    }
}

// ---------------- f32 -> bf16 weight conversion (with row padding) ----------
__global__ __launch_bounds__(256)
void cvt_w_bf16(const float* __restrict__ w, __bf16* __restrict__ o,
                long nRealRows, long nPadRows, long K) {
    const long total = nPadRows * K;
    for (long i = (long)blockIdx.x * 256 + threadIdx.x; i < total;
         i += (long)gridDim.x * 256) {
        long n = i / K, k = i - n * K;
        o[i] = (__bf16)((n < nRealRows) ? w[n * K + k] : 0.0f);
    }
}

// ---------------- bf16 WMMA GEMM:  out[M,Nreal] = A[M,K] * Bw[N,K]^T ------
// Block tile 128x256, 8 waves, 64x64 per wave, double-buffered LDS.
// EPILOGUE: 0 = f32 store, 1 = +bias, GELU, bf16 store, 2 = +bias+add0+add1, f32 store
template <int EPILOGUE>
__global__ __launch_bounds__(256)
void gemm_bf16_wmma(const __bf16* __restrict__ A,     // M x K, row-major
                    const __bf16* __restrict__ Bw,    // Npad x K, row-major
                    int M, int Npad, int K, int Nreal,
                    float*  __restrict__ Cout,
                    __bf16* __restrict__ CoutBf,
                    const float* __restrict__ bias,
                    const float* __restrict__ add0,
                    const float* __restrict__ add1) {
    __shared__ __bf16 As[2][128][40];   // 80B rows: 16B-aligned frag loads
    __shared__ __bf16 Bs[2][256][40];   // 60 KB total of 320 KB WGP LDS

    const int tid   = threadIdx.x;
    const int lane  = tid & 31;
    const int wave  = tid >> 5;
    const int wm    = wave & 1;          // 2 waves along M (64 rows each)
    const int wn    = wave >> 1;         // 4 waves along N (64 cols each)
    const int ml    = lane & 15;         // fragment row / col within 16
    const int khalf = (lane >> 4) * 8;   // K-phase: 0 or 8 halves
    const long blockM = (long)blockIdx.y * 128;
    const long blockN = (long)blockIdx.x * 256;

    v8f acc[4][4];
#pragma unroll
    for (int i = 0; i < 4; ++i)
#pragma unroll
        for (int j = 0; j < 4; ++j)
            acc[i][j] = (v8f){0.f,0.f,0.f,0.f,0.f,0.f,0.f,0.f};

    const int arow = tid >> 2;           // 0..63 (staging row base)
    const int acg  = (tid & 3) * 8;      // half-element group (16B)

    // prologue: stage K-tile 0 directly into buffer 0
#pragma unroll
    for (int it = 0; it < 2; ++it) {
        const int row = arow + it * 64;
        *(v8bf*)&As[0][row][acg] = *(const v8bf*)&A[(blockM + row) * (long)K + acg];
    }
#pragma unroll
    for (int it = 0; it < 4; ++it) {
        const int row = arow + it * 64;
        *(v8bf*)&Bs[0][row][acg] = *(const v8bf*)&Bw[(blockN + row) * (long)K + acg];
    }
    __syncthreads();

    int buf = 0;
    for (int k0 = 0; k0 < K; k0 += 32, buf ^= 1) {
        const int kn = k0 + 32;
        const bool hasNext = kn < K;

        // issue next K-tile global loads into registers (overlaps with WMMA)
        v8bf ta0, ta1, tb0, tb1, tb2, tb3;
        if (hasNext) {
            ta0 = *(const v8bf*)&A [(blockM + arow      ) * (long)K + kn + acg];
            ta1 = *(const v8bf*)&A [(blockM + arow +  64) * (long)K + kn + acg];
            tb0 = *(const v8bf*)&Bw[(blockN + arow      ) * (long)K + kn + acg];
            tb1 = *(const v8bf*)&Bw[(blockN + arow +  64) * (long)K + kn + acg];
            tb2 = *(const v8bf*)&Bw[(blockN + arow + 128) * (long)K + kn + acg];
            tb3 = *(const v8bf*)&Bw[(blockN + arow + 192) * (long)K + kn + acg];
            if (kn + 32 < K) {   // CDNA5 global_prefetch_b8 two tiles ahead
                __builtin_prefetch(&A [(blockM + arow) * (long)K + kn + 32], 0, 1);
                __builtin_prefetch(&Bw[(blockN + arow) * (long)K + kn + 32], 0, 1);
            }
        }

        // compute 16 WMMAs from current buffer (1 ds_load_b128 pair per frag)
        v16bf bfr[4];
#pragma unroll
        for (int tn = 0; tn < 4; ++tn)
            bfr[tn] = ldfrag(&Bs[buf][wn * 64 + tn * 16 + ml][khalf]);
#pragma unroll
        for (int tm = 0; tm < 4; ++tm) {
            v16bf af = ldfrag(&As[buf][wm * 64 + tm * 16 + ml][khalf]);
#pragma unroll
            for (int tn = 0; tn < 4; ++tn)
                acc[tm][tn] = __builtin_amdgcn_wmma_f32_16x16x32_bf16(
                    false, af, false, bfr[tn], (short)0, acc[tm][tn],
                    false, false);
        }

        // commit staged registers into the other buffer
        if (hasNext) {
            *(v8bf*)&As[buf ^ 1][arow      ][acg] = ta0;
            *(v8bf*)&As[buf ^ 1][arow +  64][acg] = ta1;
            *(v8bf*)&Bs[buf ^ 1][arow      ][acg] = tb0;
            *(v8bf*)&Bs[buf ^ 1][arow +  64][acg] = tb1;
            *(v8bf*)&Bs[buf ^ 1][arow + 128][acg] = tb2;
            *(v8bf*)&Bs[buf ^ 1][arow + 192][acg] = tb3;
        }
        __syncthreads();   // one barrier per K-step (double buffer)
    }

    // epilogue: VGPR r -> (M = r + 8*(lane>=16), N = lane&15)
    const int rofs = (lane >> 4) * 8;
#pragma unroll
    for (int tm = 0; tm < 4; ++tm) {
#pragma unroll
        for (int tn = 0; tn < 4; ++tn) {
            const long col = blockN + wn * 64 + tn * 16 + ml;
            if (col >= Nreal) continue;
            const long baseR = blockM + wm * 64 + tm * 16 + rofs;
#pragma unroll
            for (int r = 0; r < 8; ++r) {
                const long row = baseR + r;
                float v = acc[tm][tn][r];
                if (EPILOGUE == 0) {
                    Cout[row * (long)Nreal + col] = v;
                } else if (EPILOGUE == 1) {
                    v = gelu_tanh(v + bias[col]);
                    CoutBf[row * (long)Nreal + col] = (__bf16)v;
                } else {
                    v += bias[col] + add0[row * (long)Nreal + col]
                                   + add1[row * (long)Nreal + col];
                    Cout[row * (long)Nreal + col] = v;
                }
            }
        }
    }
}

// ---------------- SSD recurrent scan -------------------------------------
// grid: (NHEADS, BATCH), block: 256 = 64 p-lanes x 4 n-groups, 16 states/thread
__global__ __launch_bounds__(256)
void ssd_scan(const float* __restrict__ zx,
              const float* __restrict__ dt_bias,
              const float* __restrict__ A_log,
              const float* __restrict__ Dp,
              float* __restrict__ y) {
    __shared__ float sB[64], sC[64], sPart[256], sScal[2];
    const int h  = blockIdx.x;
    const int b  = blockIdx.y;
    const int kv = h >> 2;               // GQA group (REP = 4)
    const int tid = threadIdx.x;
    const int px = tid & 63;             // p lane
    const int g  = tid >> 6;             // n-group 0..3
    const float Ah    = -expf(A_log[h]);
    const float dbias = dt_bias[h];
    const float dph   = Dp[h];

    float st[16];
#pragma unroll
    for (int i = 0; i < 16; ++i) st[i] = 0.f;

    for (int t = 0; t < SEQ; ++t) {
        const long base = ((long)b * SEQ + t) * (long)NPROJ;
        if (tid < 64)        sB[tid]      = zx[base + OFF_B + kv * 64 + tid];
        else if (tid < 128)  sC[tid - 64] = zx[base + OFF_C + h  * 64 + (tid - 64)];
        else if (tid == 128) {
            float d  = zx[base + OFF_DT + h] + dbias;
            float dt = (d > 20.f) ? d : log1pf(expf(d));   // softplus
            sScal[0] = dt;
            sScal[1] = expf(dt * Ah);                       // dA
        }
        __syncthreads();
        const float dt = sScal[0];
        const float dA = sScal[1];
        const float xp = zx[base + OFF_X + kv * 64 + px];
        const float dx = dt * xp;
        float partial = 0.f;
#pragma unroll
        for (int i = 0; i < 16; ++i) {
            const int n = g * 16 + i;
            st[i] = st[i] * dA + dx * sB[n];
            partial += st[i] * sC[n];
        }
        sPart[tid] = partial;
        __syncthreads();
        if (g == 0) {
            float yv = sPart[px] + sPart[px + 64] + sPart[px + 128]
                     + sPart[px + 192] + dph * xp;
            y[((long)b * SEQ + t) * (long)DINNER + h * 64 + px] = yv;
        }
    }
}

// ---------------- gate (silu) + RMSNorm -> bf16 ---------------------------
__global__ __launch_bounds__(256)
void gate_rmsnorm(const float* __restrict__ y,
                  const float* __restrict__ zx,
                  const float* __restrict__ nw,
                  __bf16* __restrict__ out) {
    __shared__ float buf[DINNER];   // 16 KB
    __shared__ float red[256];
    const long t = blockIdx.x;
    const float* yr = y  + t * (long)DINNER;
    const float* zr = zx + t * (long)NPROJ + OFF_Z;
    float ss = 0.f;
    for (int i = threadIdx.x; i < DINNER; i += 256) {
        float z  = zr[i];
        float gv = yr[i] * (z / (1.f + expf(-z)));   // y * silu(z)
        buf[i] = gv; ss += gv * gv;
    }
    red[threadIdx.x] = ss; __syncthreads();
    for (int s = 128; s > 0; s >>= 1) {
        if (threadIdx.x < s) red[threadIdx.x] += red[threadIdx.x + s];
        __syncthreads();
    }
    const float scale = rsqrtf(red[0] * (1.0f / DINNER) + EPSV);
    for (int i = threadIdx.x; i < DINNER; i += 256)
        out[t * (long)DINNER + i] = (__bf16)(buf[i] * scale * nw[i]);
}

// ---------------- orchestration -------------------------------------------
extern "C" void kernel_launch(void* const* d_in, const int* in_sizes, int n_in,
                              void* d_out, int out_size, void* d_ws, size_t ws_size,
                              hipStream_t stream) {
    (void)in_sizes; (void)n_in; (void)out_size; (void)ws_size;

    const float* hidden    = (const float*)d_in[0];
    const float* ln_w      = (const float*)d_in[1];
    const float* ln_b      = (const float*)d_in[2];
    const float* in_proj_w = (const float*)d_in[3];
    const float* dt_bias   = (const float*)d_in[4];
    const float* A_log     = (const float*)d_in[5];
    const float* Dp        = (const float*)d_in[6];
    const float* norm_w    = (const float*)d_in[7];
    const float* outp_w    = (const float*)d_in[8];
    const float* fc1_w     = (const float*)d_in[9];
    const float* fc1_b     = (const float*)d_in[10];
    const float* fc2_w     = (const float*)d_in[11];
    const float* fc2_b     = (const float*)d_in[12];
    float* out = (float*)d_out;

    // workspace carve-up (256B aligned)
    char* w = (char*)d_ws;
    size_t off = 0;
    auto carve = [&](size_t bytes) {
        void* p = w + off;
        off = (off + bytes + 255) & ~(size_t)255;
        return p;
    };
    __bf16* h_bf      = (__bf16*)carve((size_t)NTOK * DMODEL * 2);
    __bf16* w_inproj  = (__bf16*)carve((size_t)NPROJ_PAD * DMODEL * 2);
    __bf16* w_fc1     = (__bf16*)carve((size_t)INTER * DMODEL * 2);
    __bf16* w_fc2     = (__bf16*)carve((size_t)DMODEL * INTER * 2);
    __bf16* w_outp    = (__bf16*)carve((size_t)DMODEL * DINNER * 2);
    float*  zxbcdt    = (float*)carve((size_t)NTOK * NPROJ * 4);
    float*  y_scan    = (float*)carve((size_t)NTOK * DINNER * 4);
    __bf16* yf_bf     = (__bf16*)carve((size_t)NTOK * DINNER * 2);
    __bf16* act1      = (__bf16*)carve((size_t)NTOK * INTER * 2);
    float*  mamba_out = (float*)carve((size_t)NTOK * DMODEL * 4);

    // 1) layernorm -> bf16 activations
    layernorm_to_bf16<<<NTOK, 256, 0, stream>>>(hidden, ln_w, ln_b, h_bf);

    // 2) weight conversions (grid-stride)
    cvt_w_bf16<<<4096, 256, 0, stream>>>(in_proj_w, w_inproj, NPROJ, NPROJ_PAD, DMODEL);
    cvt_w_bf16<<<4096, 256, 0, stream>>>(fc1_w,  w_fc1,  INTER,  INTER,  DMODEL);
    cvt_w_bf16<<<4096, 256, 0, stream>>>(fc2_w,  w_fc2,  DMODEL, DMODEL, INTER);
    cvt_w_bf16<<<4096, 256, 0, stream>>>(outp_w, w_outp, DMODEL, DMODEL, DINNER);

    // 3) in_proj GEMM: zxbcdt[NTOK, 10304] = h @ in_proj_w^T
    gemm_bf16_wmma<0><<<dim3(NPROJ_PAD / 256, NTOK / 128), 256, 0, stream>>>(
        h_bf, w_inproj, NTOK, NPROJ_PAD, DMODEL, NPROJ,
        zxbcdt, nullptr, nullptr, nullptr, nullptr);

    // 4) SSD scan -> y
    ssd_scan<<<dim3(NHEADS, BATCH), 256, 0, stream>>>(zxbcdt, dt_bias, A_log, Dp, y_scan);

    // 5) gate + rmsnorm -> bf16
    gate_rmsnorm<<<NTOK, 256, 0, stream>>>(y_scan, zxbcdt, norm_w, yf_bf);

    // 6) out_proj GEMM -> mamba_out (f32)
    gemm_bf16_wmma<0><<<dim3(DMODEL / 256, NTOK / 128), 256, 0, stream>>>(
        yf_bf, w_outp, NTOK, DMODEL, DINNER, DMODEL,
        mamba_out, nullptr, nullptr, nullptr, nullptr);

    // 7) fc1 GEMM + GELU -> act1 (bf16)
    gemm_bf16_wmma<1><<<dim3(INTER / 256, NTOK / 128), 256, 0, stream>>>(
        h_bf, w_fc1, NTOK, INTER, DMODEL, INTER,
        nullptr, act1, fc1_b, nullptr, nullptr);

    // 8) fc2 GEMM + bias + mamba_out + residual -> d_out (f32)
    gemm_bf16_wmma<2><<<dim3(DMODEL / 256, NTOK / 128), 256, 0, stream>>>(
        act1, w_fc2, NTOK, DMODEL, INTER, DMODEL,
        out, nullptr, fc2_b, mamba_out, hidden);
}
